// TimeConv_90812788507392
// MI455X (gfx1250) — matrix operand order repaired and checked by reading, not compiled, not run on hardware
//
#include <hip/hip_runtime.h>

// MI455X (gfx1250) implementation.
// Roofline: ~21 GFLOP of GEMMs, working set < 60 MB (L2-resident at 192MB),
// so all 2-layer MLPs go through v_wmma_f32_16x16x32_f16 (f16 in, f32 acc).
// The problem is launch/latency bound -> each level is fused to 2 kernels:
// one gather (gate+mod halves) and one 2-layer MLP whose 64x128 hidden tile
// lives in LDS (GEMM2 reads WMMA A-fragments via ds_load_b128).
// Exact simplifications: type-MLP cancels in the fanin softmax (shift
// invariance); pos-MLP folds into a 32-vector dot (posv).

#define P_NODES 8192
#define PG      4096
#define HDIM    256
#define FIN     32
#define NLVL    15

typedef __attribute__((ext_vector_type(16))) _Float16 v16h;
typedef __attribute__((ext_vector_type(8)))  _Float16 v8h;
typedef __attribute__((ext_vector_type(8)))  float    v8f;
typedef _Float16 f16;

union Frag16 { v16h v; v8h h[2]; };

// A fragment (16x32, M x K), row-major A with leading dim lda (halfs).
// ISA: lanes 0-15 = M rows, K packed {0..7,16..23}; lanes 16-31 same M,
// K packed {8..15,24..31}. Works for global or LDS pointers.
__device__ __forceinline__ v16h load_fragA(const f16* A, int lda, int row0, int k0) {
  int lane = threadIdx.x & 31;
  int m = lane & 15, hi = lane >> 4;
  const f16* p = A + (size_t)(row0 + m) * lda + k0 + hi * 8;
  Frag16 f;
  f.h[0] = *(const v8h*)(p);        // K = k0+8*hi   .. +7
  f.h[1] = *(const v8h*)(p + 16);   // K = k0+16+8*hi.. +7
  return f.v;
}

// B fragment (32x16, K x N), row-major B (K x N) leading dim ldb.
// ISA: lane L holds column N=c0+(L&15), K packed {0..7,16..23} (+8 for lanes 16-31).
__device__ __forceinline__ v16h load_fragB(const f16* B, int ldb, int k0, int c0) {
  int lane = threadIdx.x & 31;
  int n = lane & 15, hi = lane >> 4;
  const f16* p = B + (size_t)(k0 + hi * 8) * ldb + c0 + n;
  v16h v;
#pragma unroll
  for (int s = 0; s < 8; ++s) {
    v[s] = p[(size_t)s * ldb];
    v[8 + s] = p[(size_t)(16 + s) * ldb];
  }
  return v;
}

__device__ __forceinline__ v8f wmma_f16(v16h a, v16h b, v8f c) {
  return __builtin_amdgcn_wmma_f32_16x16x32_f16(false, a, false, b, (short)0, c, false, false);
}

// C/D 16x16 f32 tile: lanes 0-15 -> M = vgpr, lanes 16-31 -> M = vgpr+8, N = lane&15.
__device__ __forceinline__ void store_tile(v8f acc, float* Cf, f16* Ch, int ldc,
                                           int row, int col, const float* bias,
                                           int bcol, int act) {
  int lane = threadIdx.x & 31;
  int n = lane & 15, hi = lane >> 4;
  float bv = bias[bcol + n];
#pragma unroll
  for (int v = 0; v < 8; ++v) {
    float x = acc[v] + bv;
    if (act == 1) x = (x > 0.f) ? x : 0.1f * x;       // leaky 0.1
    else if (act == 2) x = fmaxf(x, 0.f);             // relu
    size_t o = (size_t)(row + hi * 8 + v) * ldc + col + n;
    if (Cf) Cf[o] = x; else Ch[o] = (f16)x;
  }
}

// Generic GEMM for pi/glob/out layers. Block = 8 waves; wave = 32x32 tile.
// grid = (M/256, N/32). K multiple of 32.
__global__ __launch_bounds__(256) void gemm_kernel(
    const f16* __restrict__ A, int lda,
    const f16* __restrict__ B, int ldb,
    const float* __restrict__ bias,
    float* __restrict__ Cf, f16* __restrict__ Ch, int ldc,
    int K, int rowOff, int colOff, int act) {
  int wave = threadIdx.x >> 5;
  int r0 = blockIdx.x * 256 + wave * 32;
  int c0 = blockIdx.y * 32;
  v8f acc00 = {0, 0, 0, 0, 0, 0, 0, 0};
  v8f acc01 = acc00, acc10 = acc00, acc11 = acc00;
  for (int k0 = 0; k0 < K; k0 += 32) {
    if (k0 + 32 < K)  // speculative L2 prefetch of next A K-slab
      __builtin_prefetch(A + (size_t)(r0 + (threadIdx.x & 15)) * lda + k0 + 32, 0, 1);
    v16h a0 = load_fragA(A, lda, r0,      k0);
    v16h a1 = load_fragA(A, lda, r0 + 16, k0);
    v16h b0 = load_fragB(B, ldb, k0, c0);
    v16h b1 = load_fragB(B, ldb, k0, c0 + 16);
    acc00 = wmma_f16(a0, b0, acc00);
    acc01 = wmma_f16(a0, b1, acc01);
    acc10 = wmma_f16(a1, b0, acc10);
    acc11 = wmma_f16(a1, b1, acc11);
  }
  store_tile(acc00, Cf, Ch, ldc, rowOff + r0,      colOff + c0,      bias, c0,      act);
  store_tile(acc01, Cf, Ch, ldc, rowOff + r0,      colOff + c0 + 16, bias, c0 + 16, act);
  store_tile(acc10, Cf, Ch, ldc, rowOff + r0 + 16, colOff + c0,      bias, c0,      act);
  store_tile(acc11, Cf, Ch, ldc, rowOff + r0 + 16, colOff + c0 + 16, bias, c0 + 16, act);
}

// Fused 2-layer MLP for one level half (blockIdx.z: 0=gate, 1=mod).
// Block = 8 waves = 2(M) x 4(N); owns 64 rows.
//   GEMM1: H[64x128] = leaky(X[64xK1] @ W1 + b1) -> LDS (f16)
//   GEMM2: out[64x256] = act2(H @ W2 + b2)       -> prev_h (f32)
__global__ __launch_bounds__(256) void level_mlp_kernel(
    const f16* __restrict__ Xg, const f16* __restrict__ Xm,
    const f16* __restrict__ gw1, const float* __restrict__ gb1,
    const f16* __restrict__ gw2, const float* __restrict__ gb2,
    const f16* __restrict__ mw1, const float* __restrict__ mb1,
    const f16* __restrict__ mw2, const float* __restrict__ mb2,
    float* __restrict__ outp, int act2) {
  __shared__ f16 Hs[64 * 128];
  const bool isMod  = (blockIdx.z != 0);
  const f16*   X  = isMod ? Xm  : Xg;
  const f16*   W1 = isMod ? mw1 : gw1;
  const float* B1 = isMod ? mb1 : gb1;
  const f16*   W2 = isMod ? mw2 : gw2;
  const float* B2 = isMod ? mb2 : gb2;
  const int lda    = isMod ? 320 : 288;
  const int K1     = lda;
  const int rowOut = isMod ? PG : 0;

  int wave = threadIdx.x >> 5;
  int mi = wave >> 2;          // 0..1
  int ni = wave & 3;           // 0..3
  int rblk = blockIdx.x * 64;
  int lane = threadIdx.x & 31;
  int n = lane & 15, hi = lane >> 4;

  // ---- GEMM1 into LDS ----
  {
    int r0 = rblk + mi * 32;
    int c0 = ni * 32;
    v8f a00 = {0, 0, 0, 0, 0, 0, 0, 0};
    v8f a01 = a00, a10 = a00, a11 = a00;
    for (int k0 = 0; k0 < K1; k0 += 32) {
      if (k0 + 32 < K1)
        __builtin_prefetch(X + (size_t)(r0 + (threadIdx.x & 15)) * lda + k0 + 32, 0, 1);
      v16h fa0 = load_fragA(X, lda, r0,      k0);
      v16h fa1 = load_fragA(X, lda, r0 + 16, k0);
      v16h fb0 = load_fragB(W1, 128, k0, c0);
      v16h fb1 = load_fragB(W1, 128, k0, c0 + 16);
      a00 = wmma_f16(fa0, fb0, a00);
      a01 = wmma_f16(fa0, fb1, a01);
      a10 = wmma_f16(fa1, fb0, a10);
      a11 = wmma_f16(fa1, fb1, a11);
    }
    v8f tiles[4] = {a00, a01, a10, a11};
#pragma unroll
    for (int ti = 0; ti < 4; ++ti) {
      int lr = mi * 32 + (ti >> 1) * 16 + hi * 8;
      int lc = c0 + (ti & 1) * 16 + n;
      float bv = B1[lc];
#pragma unroll
      for (int v = 0; v < 8; ++v) {
        float x = tiles[ti][v] + bv;
        x = (x > 0.f) ? x : 0.1f * x;  // leaky
        Hs[(lr + v) * 128 + lc] = (f16)x;
      }
    }
  }
  __syncthreads();

  // ---- GEMM2 from LDS ----
  {
    int lr0 = mi * 32;
    int c0 = ni * 64;
    v8f acc[2][4];
#pragma unroll
    for (int i = 0; i < 2; ++i)
#pragma unroll
      for (int j = 0; j < 4; ++j) acc[i][j] = (v8f){0, 0, 0, 0, 0, 0, 0, 0};
    for (int k0 = 0; k0 < 128; k0 += 32) {
      v16h fa0 = load_fragA(Hs, 128, lr0,      k0);  // ds_load path
      v16h fa1 = load_fragA(Hs, 128, lr0 + 16, k0);
#pragma unroll
      for (int j = 0; j < 4; ++j) {
        v16h fb = load_fragB(W2, 256, k0, c0 + j * 16);
        acc[0][j] = wmma_f16(fa0, fb, acc[0][j]);
        acc[1][j] = wmma_f16(fa1, fb, acc[1][j]);
      }
    }
#pragma unroll
    for (int i = 0; i < 2; ++i)
#pragma unroll
      for (int j = 0; j < 4; ++j)
        store_tile(acc[i][j], outp, nullptr, 256,
                   rowOut + rblk + lr0 + i * 16, c0 + j * 16, B2, c0 + j * 16, act2);
  }
}

// Fused per-level gather: blocks [0,PG) = gate half, [PG,2*PG) = mod half.
// One block per node, thread = channel.
__global__ __launch_bounds__(256) void gather_kernel(
    const float* __restrict__ prevh, const int* __restrict__ src,
    const float* __restrict__ featl, const float* __restrict__ bpl,
    const float* __restrict__ avs, const float* __restrict__ posv,
    const float* __restrict__ pw1, const float* __restrict__ pb1,
    f16* __restrict__ Xg, f16* __restrict__ Xm) {
  __shared__ int   idx[8];
  __shared__ float bps[8];
  __shared__ float sc[8];
  __shared__ float red[8 * 256];
  int t = threadIdx.x;

  if (blockIdx.x < PG) {
    // ---- gate half: per-channel fanin softmax self-attention ----
    int p = blockIdx.x;
    if (t < 8) idx[t] = src[p * 8 + t];
    __syncthreads();
    float v[8], mx = -1e30f;
#pragma unroll
    for (int f = 0; f < 8; ++f) {
      v[f] = prevh[(size_t)idx[f] * HDIM + t];
      mx = fmaxf(mx, v[f]);
    }
    float s = 0.f, acc = 0.f;
#pragma unroll
    for (int f = 0; f < 8; ++f) {
      float e = __expf(v[f] - mx);
      s += e;
      acc = fmaf(e, v[f], acc);
    }
    f16* row = Xg + (size_t)p * 288;
    row[t] = (f16)(acc / s);
    if (t < FIN) row[256 + t] = (f16)featl[(size_t)p * FIN + t];
  } else {
    // ---- mod half: logits s_f = sm_f.av_s + posScore(bp_f); softmax over fanin ----
    int p = blockIdx.x - PG;           // local row in Xm
    const int*   srcm = src + (size_t)PG * 8;
    const float* fm   = featl + (size_t)PG * FIN;
    const float* bpm  = bpl + (size_t)PG * 8;
    if (t < 8) { idx[t] = srcm[p * 8 + t]; bps[t] = bpm[p * 8 + t]; }
    __syncthreads();
    float v[8];
    float a = avs[t];
#pragma unroll
    for (int f = 0; f < 8; ++f) {
      v[f] = prevh[(size_t)idx[f] * HDIM + t];
      red[f * 256 + t] = v[f] * a;
    }
    __syncthreads();
    for (int st = 128; st >= 1; st >>= 1) {
      if (t < st) {
#pragma unroll
        for (int f = 0; f < 8; ++f) red[f * 256 + t] += red[f * 256 + t + st];
      }
      __syncthreads();
    }
    if (t < 8) {
      float b = bps[t];
      float s = posv[32];
      for (int j = 0; j < 32; ++j) {
        float h = fmaf(b, pw1[j], pb1[j]);
        h = (h > 0.f) ? h : 0.1f * h;
        s = fmaf(h, posv[j], s);
      }
      sc[t] = red[t * 256] + s;
    }
    __syncthreads();
    float mx = -1e30f;
#pragma unroll
    for (int f = 0; f < 8; ++f) mx = fmaxf(mx, sc[f]);
    float al[8], ssum = 0.f;
#pragma unroll
    for (int f = 0; f < 8; ++f) { al[f] = __expf(sc[f] - mx); ssum += al[f]; }
    float inv = 1.f / ssum;
    float nm = 0.f;
#pragma unroll
    for (int f = 0; f < 8; ++f) nm = fmaf(al[f] * inv, v[f], nm);
    f16* row = Xm + (size_t)p * 320;
    row[t] = (f16)nm;
    if (t == 0) {
      float nb = 0.f;
      for (int f = 0; f < 8; ++f) nb += al[f] * inv * bps[f];
      row[256] = (f16)nb;
    }
    if (t < FIN) row[257 + t] = (f16)fm[(size_t)p * FIN + t];
    if (t >= 32 && t < 63) row[257 + t] = (f16)0.f;   // pad cols 289..319
  }
}

__device__ void cvt_pad(const float* W, f16* T, int KN, int KpadN, int tid, int nth) {
  for (int i = tid; i < KpadN; i += nth)
    T[i] = (f16)(i < KN ? W[i] : 0.f);
}

// Weight conversion to f16 (row-major K x N unchanged; mod_w1 padded 289->320
// rows with zeros). posv[0:32] = pos_w2 @ av_p; posv[32] = pos_b2 . av_p.
__global__ void prep_kernel(const float* pi_w2, const float* gate_w1, const float* gate_w2,
                            const float* mod_w1, const float* mod_w2, const float* glob_w2,
                            const float* out_w1, const float* pos_w2, const float* pos_b2,
                            const float* attn,
                            f16* c_pi_w2, f16* c_gate_w1, f16* c_gate_w2,
                            f16* c_mod_w1, f16* c_mod_w2, f16* c_glob_w2,
                            f16* c_out_w1, float* posv) {
  int tid = blockIdx.x * blockDim.x + threadIdx.x;
  int nth = gridDim.x * blockDim.x;
  cvt_pad(pi_w2,   c_pi_w2,   128 * 256, 128 * 256, tid, nth);
  cvt_pad(gate_w1, c_gate_w1, 288 * 128, 288 * 128, tid, nth);
  cvt_pad(gate_w2, c_gate_w2, 128 * 256, 128 * 256, tid, nth);
  cvt_pad(mod_w1,  c_mod_w1,  289 * 128, 320 * 128, tid, nth);
  cvt_pad(mod_w2,  c_mod_w2,  128 * 256, 128 * 256, tid, nth);
  cvt_pad(glob_w2, c_glob_w2, 128 * 256, 128 * 256, tid, nth);
  cvt_pad(out_w1,  c_out_w1,  512 * 256, 512 * 256, tid, nth);
  if (tid < 32) {
    float s = 0.f;
    for (int nn = 0; nn < 32; ++nn) s += pos_w2[tid * 32 + nn] * attn[32 + nn];
    posv[tid] = s;
  } else if (tid == 32) {
    float s = 0.f;
    for (int nn = 0; nn < 32; ++nn) s += pos_b2[nn] * attn[32 + nn];
    posv[32] = s;
  }
}

// First layer of pi/glob MLPs (input dim 1): H[p][j] = leaky(x[p]*w1[j]+b1[j]).
__global__ void scalar_mlp1_kernel(const float* __restrict__ x, const float* __restrict__ w1,
                                   const float* __restrict__ b1, f16* __restrict__ H) {
  int i = blockIdx.x * blockDim.x + threadIdx.x;  // over P*128
  int p = i >> 7, j = i & 127;
  float v = fmaf(x[p], w1[j], b1[j]);
  v = (v > 0.f) ? v : 0.1f * v;
  H[i] = (f16)v;
}

// prev_h (P x 256 f32) -> Xout cols [0:256) f16 (row stride 512).
__global__ void copy_h_kernel(const float* __restrict__ prevh, f16* __restrict__ Xout) {
  int i = blockIdx.x * blockDim.x + threadIdx.x;  // over P*256
  int p = i >> 8, c = i & 255;
  Xout[(size_t)p * 512 + c] = (f16)prevh[i];
}

// out[p] = Hout[p] . out_w2 + out_b2
__global__ void final_kernel(const f16* __restrict__ Hout, const float* __restrict__ w2,
                             const float* __restrict__ b2, float* __restrict__ out) {
  int p = blockIdx.x * blockDim.x + threadIdx.x;
  const f16* r = Hout + (size_t)p * 256;
  float s = b2[0];
  for (int c = 0; c < 256; ++c) s = fmaf((float)r[c], w2[c], s);
  out[p] = s;
}

extern "C" void kernel_launch(void* const* d_in, const int* in_sizes, int n_in,
                              void* d_out, int out_size, void* d_ws, size_t ws_size,
                              hipStream_t stream) {
  (void)in_sizes; (void)n_in; (void)out_size; (void)ws_size;
  const float* delay   = (const float*)d_in[0];
  const float* feat    = (const float*)d_in[1];
  const float* bit_pos = (const float*)d_in[2];
  const float* po_feat = (const float*)d_in[3];
  const int*   src_idx = (const int*)d_in[4];
  const float* pi_w1   = (const float*)d_in[5];
  const float* pi_b1   = (const float*)d_in[6];
  const float* pi_w2   = (const float*)d_in[7];
  const float* pi_b2   = (const float*)d_in[8];
  const float* gate_w1 = (const float*)d_in[9];
  const float* gate_b1 = (const float*)d_in[10];
  const float* gate_w2 = (const float*)d_in[11];
  const float* gate_b2 = (const float*)d_in[12];
  const float* mod_w1  = (const float*)d_in[13];
  const float* mod_b1  = (const float*)d_in[14];
  const float* mod_w2  = (const float*)d_in[15];
  const float* mod_b2  = (const float*)d_in[16];
  // 17..20: type MLP — provably cancels in the fanin softmax; unused.
  const float* pos_w1  = (const float*)d_in[21];
  const float* pos_b1  = (const float*)d_in[22];
  const float* pos_w2  = (const float*)d_in[23];
  const float* pos_b2  = (const float*)d_in[24];
  const float* attn    = (const float*)d_in[25];
  const float* glob_w1 = (const float*)d_in[26];
  const float* glob_b1 = (const float*)d_in[27];
  const float* glob_w2 = (const float*)d_in[28];
  const float* glob_b2 = (const float*)d_in[29];
  const float* out_w1  = (const float*)d_in[30];
  const float* out_b1  = (const float*)d_in[31];
  const float* out_w2  = (const float*)d_in[32];
  const float* out_b2  = (const float*)d_in[33];

  char* ws = (char*)d_ws;
  size_t off = 0;
  auto alloc = [&](size_t bytes) -> void* {
    off = (off + 255) & ~(size_t)255;
    void* p = ws + off;
    off += bytes;
    return p;
  };

  f16* c_pi_w2   = (f16*)alloc(128 * 256 * sizeof(f16));
  f16* c_gate_w1 = (f16*)alloc(288 * 128 * sizeof(f16));
  f16* c_gate_w2 = (f16*)alloc(128 * 256 * sizeof(f16));
  f16* c_mod_w1  = (f16*)alloc(320 * 128 * sizeof(f16));
  f16* c_mod_w2  = (f16*)alloc(128 * 256 * sizeof(f16));
  f16* c_glob_w2 = (f16*)alloc(128 * 256 * sizeof(f16));
  f16* c_out_w1  = (f16*)alloc(512 * 256 * sizeof(f16));
  float* posv    = (float*)alloc(64 * sizeof(float));
  float* hA      = (float*)alloc((size_t)P_NODES * HDIM * sizeof(float));
  float* hB      = (float*)alloc((size_t)P_NODES * HDIM * sizeof(float));
  f16* h1buf     = (f16*)alloc((size_t)P_NODES * 128 * sizeof(f16));   // pi, then glob
  f16* Xg        = (f16*)alloc((size_t)PG * 288 * sizeof(f16));
  f16* Xm        = (f16*)alloc((size_t)PG * 320 * sizeof(f16));
  f16* Xout      = (f16*)alloc((size_t)P_NODES * 512 * sizeof(f16));
  f16* Hout      = (f16*)alloc((size_t)P_NODES * 256 * sizeof(f16));

  // Weight conversion + posv.
  prep_kernel<<<dim3(128), dim3(256), 0, stream>>>(
      pi_w2, gate_w1, gate_w2, mod_w1, mod_w2, glob_w2, out_w1, pos_w2, pos_b2, attn,
      c_pi_w2, c_gate_w1, c_gate_w2, c_mod_w1, c_mod_w2, c_glob_w2, c_out_w1, posv);

  // prev_h = pi_mlp(delay): layer1 (dim 1 -> 128, VALU) then WMMA GEMM 128->256.
  scalar_mlp1_kernel<<<dim3(P_NODES * 128 / 256), dim3(256), 0, stream>>>(delay, pi_w1, pi_b1, h1buf);
  gemm_kernel<<<dim3(P_NODES / 256, 256 / 32), dim3(256), 0, stream>>>(
      h1buf, 128, c_pi_w2, 256, pi_b2, hA, nullptr, 256, 128, 0, 0, /*act=*/0);

  float* cur = hA;
  float* nxt = hB;
  for (int l = 0; l < NLVL; ++l) {
    const int*   srcl  = src_idx + (size_t)l * P_NODES * 8;
    const float* featl = feat + (size_t)(l + 1) * P_NODES * FIN;
    const float* bpl   = bit_pos + (size_t)l * P_NODES * 8;
    int actOut = (l < NLVL - 1) ? 2 : 0;  // relu between levels, none on last

    gather_kernel<<<dim3(2 * PG), dim3(256), 0, stream>>>(
        cur, srcl, featl, bpl, attn + 64, posv, pos_w1, pos_b1, Xg, Xm);

    level_mlp_kernel<<<dim3(PG / 64, 1, 2), dim3(256), 0, stream>>>(
        Xg, Xm, c_gate_w1, gate_b1, c_gate_w2, gate_b2,
        c_mod_w1, mod_b1, c_mod_w2, mod_b2, nxt, actOut);

    float* t = cur; cur = nxt; nxt = t;
  }

  // h_global = glob_mlp(po_feat) written as f16 into Xout cols [256:512).
  scalar_mlp1_kernel<<<dim3(P_NODES * 128 / 256), dim3(256), 0, stream>>>(po_feat, glob_w1, glob_b1, h1buf);
  gemm_kernel<<<dim3(P_NODES / 256, 256 / 32), dim3(256), 0, stream>>>(
      h1buf, 128, c_glob_w2, 256, glob_b2, nullptr, Xout, 512, 128, 0, /*colOff=*/256, /*act=*/0);
  // Xout cols [0:256) = final prev_h.
  copy_h_kernel<<<dim3(P_NODES * HDIM / 256), dim3(256), 0, stream>>>(cur, Xout);

  // out MLP layer1: (8192 x 512) -> 256 (leaky, f16)
  gemm_kernel<<<dim3(P_NODES / 256, 256 / 32), dim3(256), 0, stream>>>(
      Xout, 512, c_out_w1, 256, out_b1, nullptr, Hout, 256, 512, 0, 0, /*act=*/1);
  // out MLP layer2: 256 -> 1 dot product.
  final_kernel<<<dim3(P_NODES / 256), dim3(256), 0, stream>>>(Hout, out_w2, out_b2, (float*)d_out);
}
